// PathTransformer_47278999994619
// MI455X (gfx1250) — matrix operand order repaired and checked by reading, not compiled, not run on hardware
//
#include <hip/hip_runtime.h>
#include <hip/hip_bf16.h>

#define B_    8
#define L0_   1023
#define L_    1024
#define DIN_  16
#define D_    256
#define H_    8
#define HD_   32
#define NL_   3
#define DFF_  1024
#define NRBF_ 8
#define ALPHA_ 0.5f
#define BETA_  1.0f
#define NEG_  -100000.0f
#define M_    (B_*L_)   /* 8192 rows */

typedef __attribute__((ext_vector_type(16))) __bf16          v16bf;
typedef __attribute__((ext_vector_type(16))) unsigned short  u16x16;
typedef __attribute__((ext_vector_type(8)))  unsigned short  u16x8;
typedef __attribute__((ext_vector_type(8)))  float           v8f;

// ---------------------------------------------------------------- helpers
__device__ inline unsigned short f2bf(float f) {
    unsigned u = __builtin_bit_cast(unsigned, f);
    u += 0x7fffu + ((u >> 16) & 1u);          // round-to-nearest-even
    return (unsigned short)(u >> 16);
}

__device__ inline v16bf pack16(u16x8 lo, u16x8 hi) {
    u16x16 u;
#pragma unroll
    for (int e = 0; e < 8; ++e) { u[e] = lo[e]; u[e + 8] = hi[e]; }
    return __builtin_bit_cast(v16bf, u);
}

// A-operand fragment (16x32 bf16), tile stored row-major [m][k], row stride
// rs elements with rs*2 a multiple of 16B. Per lane the A layout is two
// CONTIGUOUS 16B chunks: k in [kh, kh+8) and [kh+16, kh+24), kh=(lane>>4)*8.
__device__ inline v16bf frag_a_lds(const unsigned short* t, int rs, int lane) {
    const unsigned short* row = t + (lane & 15) * rs + (lane >> 4) * 8;
    u16x8 lo = *(const u16x8*)(row);
    u16x8 hi = *(const u16x8*)(row + 16);
    return pack16(lo, hi);
}

// B-operand fragment (32x16 bf16), tile stored n-major [n][k] (k contiguous):
// per lane two 16B chunks: k in [kh, kh+8) and [kh+16, kh+24), kh=(lane>>4)*8.
__device__ inline v16bf frag_b_lds(const unsigned short* t, int rs, int lane) {
    const unsigned short* row = t + (lane & 15) * rs + (lane >> 4) * 8;
    u16x8 lo = *(const u16x8*)(row);
    u16x8 hi = *(const u16x8*)(row + 16);
    return pack16(lo, hi);
}

// reductions across the 16-lane half-wave (wave32, masks 1..8 stay in half)
__device__ inline float hmax16(float v) {
#pragma unroll
    for (int m = 1; m < 16; m <<= 1) v = fmaxf(v, __shfl_xor(v, m, 32));
    return v;
}
__device__ inline float hsum16(float v) {
#pragma unroll
    for (int m = 1; m < 16; m <<= 1) v += __shfl_xor(v, m, 32);
    return v;
}

// ---------------------------------------------------------------- f32 -> bf16
__global__ void f32_to_bf16_kernel(const float* __restrict__ in,
                                   unsigned short* __restrict__ out, int n) {
    int i = blockIdx.x * blockDim.x + threadIdx.x;
    if (i < n) out[i] = f2bf(in[i]);
}

// ---------------------------------------------------------------- embedding
__global__ void __launch_bounds__(256) embed_kernel(
    const float* __restrict__ x, const float* __restrict__ c_local,
    const float* __restrict__ c_sink, const int* __restrict__ lengths,
    const float* __restrict__ input_w, const float* __restrict__ input_b,
    const float* __restrict__ cls_token, const float* __restrict__ pos_emb,
    const float* __restrict__ corr_w, const float* __restrict__ corr_b,
    const float* __restrict__ rbf_c, const float* __restrict__ rbf_w,
    float* __restrict__ hbuf, float* __restrict__ clb,
    float* __restrict__ csb, float* __restrict__ mkb)
{
    __shared__ float zf[280];
    int t = blockIdx.x; int b = t / L_; int l = t - b * L_; int d = threadIdx.x;
    float clraw = 0.f, csraw = 0.f; int msk = 0;
    if (l > 0) {
        clraw = c_local[(size_t)b * L0_ + l - 1];
        csraw = c_sink [(size_t)b * L0_ + l - 1];
        msk = (l - 1) >= lengths[b];
    }
    float cl = ALPHA_ * fminf(fmaxf(clraw, 0.f), 1.f);
    float cs = BETA_  * fminf(fmaxf(csraw, 0.f), 1.f);
    if (d == 0) {
        clb[t] = clraw; csb[t] = csraw; mkb[t] = msk ? 1.f : 0.f;
        zf[256] = cl; zf[257] = cs; zf[258] = cl * cs;
        for (int i = 0; i < NRBF_; ++i) {
            float da = (cl - rbf_c[i]) / (rbf_w[i] + 1e-6f);
            zf[259 + i] = __expf(-0.5f * da * da);
            float db = (cs - rbf_c[i]) / (rbf_w[i] + 1e-6f);
            zf[267 + i] = __expf(-0.5f * db * db);
        }
    }
    zf[d] = pos_emb[(size_t)l * D_ + d];
    __syncthreads();
    float hb;
    if (l == 0) hb = cls_token[d];
    else {
        float s = input_b[d];
        const float* xr = x + ((size_t)b * L0_ + l - 1) * DIN_;
        const float* wr = input_w + (size_t)d * DIN_;
#pragma unroll
        for (int k = 0; k < DIN_; ++k) s += xr[k] * wr[k];
        hb = s;
    }
    float pe = 0.f;
    if (!msk) {
        float s = corr_b[d];
        const float* cw = corr_w + (size_t)d * 275;
        for (int k = 0; k < 275; ++k) s += zf[k] * cw[k];
        pe = s;
    }
    hbuf[(size_t)t * D_ + d] = hb + pe;
}

// ---------------------------------------------------------------- layernorm
__global__ void __launch_bounds__(256) ln_bf16_kernel(
    const float* __restrict__ xin, const float* __restrict__ g,
    const float* __restrict__ bta, unsigned short* __restrict__ out)
{
    __shared__ float red[256];
    int t = blockIdx.x, d = threadIdx.x;
    float v = xin[(size_t)t * D_ + d];
    red[d] = v; __syncthreads();
    for (int s = 128; s > 0; s >>= 1) { if (d < s) red[d] += red[d + s]; __syncthreads(); }
    float mean = red[0] * (1.0f / D_);
    __syncthreads();
    float c = v - mean;
    red[d] = c * c; __syncthreads();
    for (int s = 128; s > 0; s >>= 1) { if (d < s) red[d] += red[d + s]; __syncthreads(); }
    float var = red[0] * (1.0f / D_);
    float y = c * rsqrtf(var + 1e-5f) * g[d] + bta[d];
    out[(size_t)t * D_ + d] = f2bf(y);
}

__global__ void __launch_bounds__(256) lnout_kernel(
    const float* __restrict__ xin, const float* __restrict__ g,
    const float* __restrict__ bta, float* __restrict__ out)
{
    __shared__ float red[256];
    int b = blockIdx.x, d = threadIdx.x;
    size_t t = (size_t)b * L_;               // CLS token
    float v = xin[t * D_ + d];
    red[d] = v; __syncthreads();
    for (int s = 128; s > 0; s >>= 1) { if (d < s) red[d] += red[d + s]; __syncthreads(); }
    float mean = red[0] * (1.0f / D_);
    __syncthreads();
    float c = v - mean;
    red[d] = c * c; __syncthreads();
    for (int s = 128; s > 0; s >>= 1) { if (d < s) red[d] += red[d + s]; __syncthreads(); }
    float var = red[0] * (1.0f / D_);
    out[(size_t)b * D_ + d] = c * rsqrtf(var + 1e-5f) * g[d] + bta[d];
}

// ---------------------------------------------------------------- WMMA GEMM
// C[M,N] = A[M,K](bf16) * W[N,K]^T(bf16) + bias[N]  (+epilogues)
// 64x128 block tile, 8 waves; each wave: 1 m-tile x 4 n-tiles of 16x16
// (A-fragment reused 4x per WMMA group). Tiles are brought into LDS with
// GLOBAL_LOAD_ASYNC_TO_LDS_B128 (ASYNCcnt path), one 16B slot per thread.
template<bool GELU, bool RESID, bool STOREF, bool STOREH>
__global__ void __launch_bounds__(256) gemm_bf16_kernel(
    const unsigned short* __restrict__ A, const unsigned short* __restrict__ W,
    const float* __restrict__ bias, const float* __restrict__ resid,
    float* __restrict__ outF, unsigned short* __restrict__ outH,
    int M, int N, int K)
{
    __shared__ unsigned short sA[64 * 40];    // 80B row stride (16B aligned)
    __shared__ unsigned short sB[128 * 40];
    const int tid = threadIdx.x;
    const int lane = tid & 31, w = tid >> 5;
    const int m0 = blockIdx.y * 64, n0 = blockIdx.x * 128;
    const int mt = w >> 1, nt0 = (w & 1) * 4;
    v8f acc[4];
#pragma unroll
    for (int i = 0; i < 4; ++i) acc[i] = (v8f){0, 0, 0, 0, 0, 0, 0, 0};

    const int ra = tid >> 2, ca = (tid & 3) * 8;     // A tile: 64 rows x 4 slots
    const int rb1 = (tid + 256) >> 2;                // B tile: 128 rows, 2 slots/thread
    const unsigned ldsA  = (unsigned)(size_t)(sA + ra  * 40 + ca);
    const unsigned ldsB0 = (unsigned)(size_t)(sB + ra  * 40 + ca);
    const unsigned ldsB1 = (unsigned)(size_t)(sB + rb1 * 40 + ca);

    for (int kk = 0; kk < K; kk += 32) {
        __syncthreads();
        unsigned offA  = (unsigned)(((size_t)(m0 + ra)  * K + kk + ca) * 2);
        unsigned offB0 = (unsigned)(((size_t)(n0 + ra)  * K + kk + ca) * 2);
        unsigned offB1 = (unsigned)(((size_t)(n0 + rb1) * K + kk + ca) * 2);
        asm volatile(
            "global_load_async_to_lds_b128 %0, %1, %2\n\t"
            "global_load_async_to_lds_b128 %3, %4, %5\n\t"
            "global_load_async_to_lds_b128 %6, %7, %5\n\t"
            "s_wait_asynccnt 0"
            :
            : "v"(ldsA),  "v"(offA),  "s"(A),
              "v"(ldsB0), "v"(offB0), "s"(W),
              "v"(ldsB1), "v"(offB1)
            : "memory");
        __syncthreads();
        v16bf af = frag_a_lds(sA + mt * 16 * 40, 40, lane);
#pragma unroll
        for (int t = 0; t < 4; ++t) {
            v16bf bfr = frag_b_lds(sB + (nt0 + t) * 16 * 40, 40, lane);
            acc[t] = __builtin_amdgcn_wmma_f32_16x16x32_bf16(false, af, false, bfr,
                                                             (short)0, acc[t], false, false);
        }
    }
#pragma unroll
    for (int t = 0; t < 4; ++t) {
#pragma unroll
        for (int r = 0; r < 8; ++r) {
            int mg = m0 + mt * 16 + r + (lane >> 4) * 8;
            int ng = n0 + (nt0 + t) * 16 + (lane & 15);
            float v = acc[t][r] + bias[ng];
            if (GELU)  v = 0.5f * v * (1.0f + erff(v * 0.70710678118f));
            if (RESID) v += resid[(size_t)mg * N + ng];
            if (STOREF) outF[(size_t)mg * N + ng] = v;
            if (STOREH) outH[(size_t)mg * N + ng] = f2bf(v);
        }
    }
}

// ---------------------------------------------------------------- fused attention
// One workgroup per (b, h, 128-row block); wave w owns rows [base+16w, +16).
// Flash-style online softmax, bias computed analytically (neighbor + rank-1).
// K chunk is copied via async-to-LDS; V is transposed element-wise into LDS.
__global__ void __launch_bounds__(256) attn_kernel(
    const unsigned short* __restrict__ qkv, const float* __restrict__ clv,
    const float* __restrict__ csv, const float* __restrict__ mk,
    unsigned short* __restrict__ out)
{
    __shared__ unsigned short sK[32 * 40];       // [token][dim], 16B-aligned rows
    __shared__ unsigned short sV[32 * 40];       // transposed: [dim][token]
    __shared__ unsigned short sP[8][16 * 40];    // per-wave P staging
    const int tid = threadIdx.x, lane = tid & 31, w = tid >> 5;
    const int b = blockIdx.z, h = blockIdx.y;
    const int rowbase = blockIdx.x * 128 + w * 16;
    const size_t base = (size_t)b * L_;
    const float scale = 0.17677669529663689f;    // 1/sqrt(32)

    // Q fragment straight from global (A-operand layout, two aligned 16B runs)
    v16bf qf;
    {
        const unsigned short* qrow =
            qkv + (base + rowbase + (lane & 15)) * 768 + h * HD_ + (lane >> 4) * 8;
        u16x8 lo = *(const u16x8*)(qrow);
        u16x8 hi = *(const u16x8*)(qrow + 16);
        qf = pack16(lo, hi);
    }

    float rmax[8], rsum[8], csi[8], cli[8]; int mi[8];
#pragma unroll
    for (int r = 0; r < 8; ++r) {
        int i = rowbase + r + (lane >> 4) * 8;
        rmax[r] = -3.0e38f; rsum[r] = 0.0f;
        csi[r] = csv[base + i]; cli[r] = clv[base + i];
        mi[r] = (mk[base + i] != 0.0f);
    }
    v8f o0 = {0, 0, 0, 0, 0, 0, 0, 0};
    v8f o1 = {0, 0, 0, 0, 0, 0, 0, 0};
    const int ldr = tid >> 3, ldc = (tid & 7) * 4;
    const unsigned ldsK = (unsigned)(size_t)(sK + (tid >> 2) * 40 + (tid & 3) * 8);

    for (int jc = 0; jc < L_; jc += 32) {
        __syncthreads();
        if (tid < 128) {                          // async copy of K chunk (32x32)
            unsigned goff = (unsigned)(((base + jc + (tid >> 2)) * 768
                                        + 256 + h * HD_ + (tid & 3) * 8) * 2);
            asm volatile(
                "global_load_async_to_lds_b128 %0, %1, %2\n\t"
                "s_wait_asynccnt 0"
                :
                : "v"(ldsK), "v"(goff), "s"(qkv)
                : "memory");
        }
        {   // V chunk, transposed for B-operand reads
            const unsigned short* vg = qkv + (base + jc + ldr) * 768 + 512 + h * HD_ + ldc;
#pragma unroll
            for (int i = 0; i < 4; ++i)
                sV[(ldc + i) * 40 + ldr] = vg[i];
        }
        __syncthreads();
        v16bf kf0 = frag_b_lds(sK, 40, lane);
        v16bf kf1 = frag_b_lds(sK + 16 * 40, 40, lane);
        v8f z = {0, 0, 0, 0, 0, 0, 0, 0};
        v8f s0 = __builtin_amdgcn_wmma_f32_16x16x32_bf16(false, qf, false, kf0,
                                                         (short)0, z, false, false);
        v8f s1 = __builtin_amdgcn_wmma_f32_16x16x32_bf16(false, qf, false, kf1,
                                                         (short)0, z, false, false);
        int j0 = jc + (lane & 15), j1 = j0 + 16;
        float csj0 = csv[base + j0], csj1 = csv[base + j1];
        float clj0 = clv[base + j0], clj1 = clv[base + j1];
        int mj0 = (mk[base + j0] != 0.0f), mj1 = (mk[base + j1] != 0.0f);
        unsigned short* myP = sP[w];
#pragma unroll
        for (int r = 0; r < 8; ++r) {
            int i = rowbase + r + (lane >> 4) * 8;
            float a0 = s0[r] * scale + BETA_ * csi[r] * csj0;
            float a1 = s1[r] * scale + BETA_ * csi[r] * csj1;
            int d0 = i - j0, d1 = i - j1;
            if (d0 == 1 || d0 == -1)
                a0 += ALPHA_ * ((i == L_ - 1 && j0 == L_ - 2) ? cli[r] : clj0);
            if (d1 == 1 || d1 == -1)
                a1 += ALPHA_ * ((i == L_ - 1 && j1 == L_ - 2) ? cli[r] : clj1);
            if (mi[r] || mj0) a0 = NEG_;
            if (mi[r] || mj1) a1 = NEG_;
            float ml = hmax16(fmaxf(a0, a1));
            float mn = fmaxf(rmax[r], ml);
            float corr = __expf(rmax[r] - mn);
            float p0 = __expf(a0 - mn), p1 = __expf(a1 - mn);
            rsum[r] = rsum[r] * corr + hsum16(p0 + p1);
            rmax[r] = mn;
            o0[r] *= corr; o1[r] *= corr;
            int m = r + (lane >> 4) * 8;          // C-layout -> LDS transpose
            myP[m * 40 + (lane & 15)] = f2bf(p0);
            myP[m * 40 + 16 + (lane & 15)] = f2bf(p1);
        }
        asm volatile("s_wait_dscnt 0" ::: "memory");  // in-wave LDS RAW fence
        v16bf pf  = frag_a_lds(myP, 40, lane);
        v16bf vf0 = frag_b_lds(sV, 40, lane);
        v16bf vf1 = frag_b_lds(sV + 16 * 40, 40, lane);
        o0 = __builtin_amdgcn_wmma_f32_16x16x32_bf16(false, pf, false, vf0,
                                                     (short)0, o0, false, false);
        o1 = __builtin_amdgcn_wmma_f32_16x16x32_bf16(false, pf, false, vf1,
                                                     (short)0, o1, false, false);
    }
#pragma unroll
    for (int r = 0; r < 8; ++r) {
        int tok = rowbase + r + (lane >> 4) * 8;
        float inv = 1.0f / rsum[r];
        int dim0 = h * HD_ + (lane & 15);
        out[(base + tok) * D_ + dim0]      = f2bf(o0[r] * inv);
        out[(base + tok) * D_ + dim0 + 16] = f2bf(o1[r] * inv);
    }
}

// ---------------------------------------------------------------- launch
extern "C" void kernel_launch(void* const* d_in, const int* in_sizes, int n_in,
                              void* d_out, int out_size, void* d_ws, size_t ws_size,
                              hipStream_t stream)
{
    (void)in_sizes; (void)n_in; (void)out_size; (void)ws_size;
    const float* x         = (const float*)d_in[0];
    const float* c_local   = (const float*)d_in[1];
    const float* c_sink    = (const float*)d_in[2];
    const int*   lengths   = (const int*)  d_in[3];
    const float* input_w   = (const float*)d_in[4];
    const float* input_b   = (const float*)d_in[5];
    const float* cls_tok   = (const float*)d_in[6];
    const float* pos_emb   = (const float*)d_in[7];
    const float* corr_w    = (const float*)d_in[8];
    const float* corr_b    = (const float*)d_in[9];
    const float* rbf_c     = (const float*)d_in[10];
    const float* rbf_w     = (const float*)d_in[11];
    const float* in_proj_w = (const float*)d_in[12];
    const float* in_proj_b = (const float*)d_in[13];
    const float* out_w     = (const float*)d_in[14];
    const float* out_b     = (const float*)d_in[15];
    const float* ff_w1     = (const float*)d_in[16];
    const float* ff_b1     = (const float*)d_in[17];
    const float* ff_w2     = (const float*)d_in[18];
    const float* ff_b2     = (const float*)d_in[19];
    const float* ln1_g     = (const float*)d_in[20];
    const float* ln1_b     = (const float*)d_in[21];
    const float* ln2_g     = (const float*)d_in[22];
    const float* ln2_b     = (const float*)d_in[23];
    const float* lnout_g   = (const float*)d_in[24];
    const float* lnout_b   = (const float*)d_in[25];

    char* p = (char*)d_ws;
    auto carve = [&](size_t bytes) -> void* {
        void* r = (void*)p; p += (bytes + 255) & ~(size_t)255; return r;
    };
    float*          hbuf   = (float*)         carve((size_t)M_ * D_   * 4);
    unsigned short* ybf    = (unsigned short*)carve((size_t)M_ * D_   * 2);
    unsigned short* qkvbf  = (unsigned short*)carve((size_t)M_ * 768  * 2);
    unsigned short* attnbf = (unsigned short*)carve((size_t)M_ * D_   * 2);
    unsigned short* gbf    = (unsigned short*)carve((size_t)M_ * DFF_ * 2);
    unsigned short* wqkv   = (unsigned short*)carve((size_t)NL_ * 768 * D_  * 2);
    unsigned short* wout   = (unsigned short*)carve((size_t)NL_ * D_  * D_  * 2);
    unsigned short* wff1   = (unsigned short*)carve((size_t)NL_ * DFF_* D_  * 2);
    unsigned short* wff2   = (unsigned short*)carve((size_t)NL_ * D_  * DFF_* 2);
    float*          clb    = (float*)         carve((size_t)M_ * 4);
    float*          csb    = (float*)         carve((size_t)M_ * 4);
    float*          mkb    = (float*)         carve((size_t)M_ * 4);

    auto cvt = [&](const float* src, unsigned short* dst, int n) {
        f32_to_bf16_kernel<<<(n + 255) / 256, 256, 0, stream>>>(src, dst, n);
    };
    cvt(in_proj_w, wqkv, NL_ * 768 * D_);
    cvt(out_w,     wout, NL_ * D_ * D_);
    cvt(ff_w1,     wff1, NL_ * DFF_ * D_);
    cvt(ff_w2,     wff2, NL_ * D_ * DFF_);

    embed_kernel<<<M_, 256, 0, stream>>>(x, c_local, c_sink, lengths, input_w,
        input_b, cls_tok, pos_emb, corr_w, corr_b, rbf_c, rbf_w,
        hbuf, clb, csb, mkb);

    for (int l = 0; l < NL_; ++l) {
        ln_bf16_kernel<<<M_, 256, 0, stream>>>(hbuf, ln1_g + l * D_, ln1_b + l * D_, ybf);
        gemm_bf16_kernel<false,false,false,true><<<dim3(768 / 128, M_ / 64), 256, 0, stream>>>(
            ybf, wqkv + (size_t)l * 768 * D_, in_proj_b + l * 768,
            nullptr, nullptr, qkvbf, M_, 768, D_);
        attn_kernel<<<dim3(L_ / 128, H_, B_), 256, 0, stream>>>(qkvbf, clb, csb, mkb, attnbf);
        gemm_bf16_kernel<false,true,true,false><<<dim3(D_ / 128, M_ / 64), 256, 0, stream>>>(
            attnbf, wout + (size_t)l * D_ * D_, out_b + l * D_,
            hbuf, hbuf, nullptr, M_, D_, D_);
        ln_bf16_kernel<<<M_, 256, 0, stream>>>(hbuf, ln2_g + l * D_, ln2_b + l * D_, ybf);
        gemm_bf16_kernel<true,false,false,true><<<dim3(DFF_ / 128, M_ / 64), 256, 0, stream>>>(
            ybf, wff1 + (size_t)l * DFF_ * D_, ff_b1 + l * DFF_,
            nullptr, nullptr, gbf, M_, DFF_, D_);
        gemm_bf16_kernel<false,true,true,false><<<dim3(D_ / 128, M_ / 64), 256, 0, stream>>>(
            gbf, wff2 + (size_t)l * D_ * DFF_, ff_b2 + l * D_,
            hbuf, hbuf, nullptr, M_, D_, DFF_);
    }
    lnout_kernel<<<B_, 256, 0, stream>>>(hbuf, lnout_g, lnout_b, (float*)d_out);
}
